// NeuralTuringMachine_47708496724401
// MI455X (gfx1250) — matrix operand order repaired and checked by reading, not compiled, not run on hardware
//
#include <hip/hip_runtime.h>
#include <hip/hip_bf16.h>
#include <math.h>

// ---------------------------------------------------------------------------
// NTM forward, MI455X (gfx1250, wave32).
// GEMMs: v_wmma_f32_16x16x32_bf16 + global_load_async_to_lds_b128 (ASYNCcnt)
// with triple-buffered LDS tiles, one barrier per K-step.
// dims: B=128 S=32 IN=512 H=1024 OUT=512 N=16384 W=128 C=899
// ---------------------------------------------------------------------------

typedef __attribute__((ext_vector_type(16))) __bf16 v16bf;
typedef __attribute__((ext_vector_type(8)))  __bf16 v8bf;
typedef __attribute__((ext_vector_type(8)))  float  v8f;

#define DEVI static __device__ __forceinline__

DEVI unsigned short f2bf(float f) {
  union { float f; unsigned u; } v; v.f = f;
  unsigned u = v.u;
  u += 0x7FFFu + ((u >> 16) & 1u);   // round-to-nearest-even
  return (unsigned short)(u >> 16);
}
DEVI float sigf(float x) { return 1.f / (1.f + __expf(-x)); }
DEVI float softplusf(float x) { return x > 20.f ? x : log1pf(__expf(x)); }

// gfx1250: generic LDS pointer = {shared_aperture[31:0]_hi, lds_offset[31:0]},
// so the low 32 bits are exactly the wave-relative LDS address VDST wants.
DEVI unsigned lds_addr32(const void* p) { return (unsigned)(uintptr_t)p; }

// Async copy 16B global -> LDS, tracked by ASYNCcnt (per-lane addresses).
DEVI void async_cp16(const unsigned short* gptr, unsigned short* lptr) {
  unsigned loff = lds_addr32(lptr);
  asm volatile("global_load_async_to_lds_b128 %0, %1, off"
               :: "v"(loff), "v"(gptr) : "memory");
}

// ---------------------------------------------------------------------------
// WMMA GEMM:  C[M,N] = A[M,K] * Bt[N,K]^T (+bias[N])
// A, Bt bf16 (as ushort) row-major, leading dims lda/ldb (elements).
// Requires: M %128==0, N %64==0, K %32==0, lda/ldb %8==0, 16B-aligned bases.
// Block tile 128x64, 256 threads (8 waves); wave = 16 rows x 64 cols
// (1 A-frag, 4 B-frags, 4 WMMAs per 32-wide K-step). Triple-buffered LDS.
// ---------------------------------------------------------------------------
__global__ __launch_bounds__(256)
void gemm_bf16_nt(const unsigned short* __restrict__ A,
                  const unsigned short* __restrict__ Bt,
                  const float* __restrict__ bias,
                  float* __restrict__ C,
                  int M, int N, int K, int lda, int ldb, int ldc)
{
  __shared__ unsigned short As[3][128 * 40];
  __shared__ unsigned short Bs[3][64 * 40];

  const int tid  = threadIdx.x;
  const int lane = tid & 31;
  const int wid  = tid >> 5;
  const int m0   = blockIdx.y << 7;
  const int n0   = blockIdx.x << 6;
  const int wrow = wid << 4;                // 16-row band per wave

  // cooperative async-copy mapping: thread -> (row tid>>2, seg (tid&3)*8)
  const int arow = tid >> 2;                // 0..63
  const int aseg = (tid & 3) << 3;          // 0,8,16,24 elements

  // fragment mapping (ISA 7.12.2)
  const int mrow  = lane & 15;
  const int khalf = (lane >> 4) << 3;       // A K-piece base: 0 or 8
  const int kb    = (lane >> 4) << 3;       // B K-piece base: 0 or 8

  const int nk = K >> 5;

  auto issue_tile = [&](int buf, int k0) {
    async_cp16(&A [(size_t)(m0 + arow)      * lda + k0 + aseg], &As[buf][ arow       * 40 + aseg]);
    async_cp16(&A [(size_t)(m0 + arow + 64) * lda + k0 + aseg], &As[buf][(arow + 64) * 40 + aseg]);
    async_cp16(&Bt[(size_t)(n0 + arow)      * ldb + k0 + aseg], &Bs[buf][ arow       * 40 + aseg]);
  };

  v8f acc[4];
#pragma unroll
  for (int cb = 0; cb < 4; ++cb) acc[cb] = (v8f){};

  // prologue: tiles 0 and 1 in flight (3 async ops per wave per tile)
  issue_tile(0, 0);
  issue_tile(1, (nk > 1) ? 32 : 0);

  for (int t = 0; t < nk; ++t) {
    // retire tile t (leave the next tile's 3 ops outstanding), then sync waves
    asm volatile("s_wait_asynccnt 0x3" ::: "memory");
    __syncthreads();

    // keep two tiles ahead; tail issues clamped re-reads into a dead buffer
    int tn = t + 2; if (tn >= nk) tn = nk - 1;
    issue_tile((t + 2) % 3, tn << 5);

    const unsigned short* as = &As[t % 3][0];
    const unsigned short* bs = &Bs[t % 3][0];

    // A fragment: row (wrow+mrow); K = {khalf..+7, khalf+16..+23}
    const unsigned short* ap = &as[(wrow + mrow) * 40];
    v8bf alo = *(const v8bf*)(ap + khalf);
    v8bf ahi = *(const v8bf*)(ap + khalf + 16);
    v16bf afrag;
#pragma unroll
    for (int i = 0; i < 8; ++i) { afrag[i] = alo[i]; afrag[i + 8] = ahi[i]; }

#pragma unroll
    for (int cb = 0; cb < 4; ++cb) {
      const unsigned short* bp = &bs[(cb * 16 + mrow) * 40 + kb];
      v8bf blo = *(const v8bf*)(bp);
      v8bf bhi = *(const v8bf*)(bp + 16);
      v16bf bfrag;
#pragma unroll
      for (int i = 0; i < 8; ++i) { bfrag[i] = blo[i]; bfrag[i + 8] = bhi[i]; }
      acc[cb] = __builtin_amdgcn_wmma_f32_16x16x32_bf16(false, afrag, false, bfrag,
                                                        (short)0, acc[cb], false, false);
    }
  }

  // C/D layout: lane holds col = lane%16; VGPR r holds row r + 8*(lane/16)
  const int row0 = m0 + wrow + ((lane >> 4) << 3);
#pragma unroll
  for (int cb = 0; cb < 4; ++cb) {
    const int col = n0 + cb * 16 + (lane & 15);
    const float bv = bias ? bias[col] : 0.f;
#pragma unroll
    for (int r = 0; r < 8; ++r)
      C[(size_t)(row0 + r) * ldc + col] = acc[cb][r] + bv;
  }
}

// ---------------------------------------------------------------------------
// Staging kernels (run once per launch; weights are constant across steps)
// ---------------------------------------------------------------------------
__global__ void stage_gates(const float* Wih, const float* Whh,
                            const float* bih, const float* bhh,
                            unsigned short* Wcat, float* bias_g)
{
  size_t idx = (size_t)blockIdx.x * blockDim.x + threadIdx.x;
  const size_t total = (size_t)4096 * 1664;
  if (idx < total) {
    int j = (int)(idx / 1664), k = (int)(idx % 1664);
    float v = (k < 640) ? Wih[(size_t)j * 640 + k] : Whh[(size_t)j * 1024 + (k - 640)];
    Wcat[idx] = f2bf(v);
  }
  if (idx < 4096) bias_g[idx] = bih[idx] + bhh[idx];
}

__global__ void stage_wout(const float* Wout, const float* bout,
                           unsigned short* Wb, float* bb)
{
  size_t idx = (size_t)blockIdx.x * blockDim.x + threadIdx.x;
  const size_t total = (size_t)960 * 1024;
  if (idx < total) {
    int n = (int)(idx / 1024), k = (int)(idx % 1024);
    Wb[idx] = f2bf(n < 899 ? Wout[(size_t)n * 1024 + k] : 0.f);
  }
  if (idx < 960) bb[idx] = (idx < 899) ? bout[idx] : 0.f;
}

__global__ void stage_heads(const float* rkW, const float* rkb,
                            const float* rbW, const float* rbb,
                            const float* rgW, const float* rgb,
                            const float* wkW, const float* wkb,
                            const float* wbW, const float* wbb,
                            const float* wgW, const float* wgb,
                            const float* erW, const float* erb,
                            const float* adW, const float* adb,
                            unsigned short* HWb, float* hbias)
{
  size_t idx = (size_t)blockIdx.x * blockDim.x + threadIdx.x;
  const size_t total = (size_t)576 * 928;
  if (idx < total) {
    int n = (int)(idx / 928), k = (int)(idx % 928);
    float v = 0.f;
    if (k < 899) {
      if      (n < 128)  v = rkW[(size_t)n * 899 + k];
      else if (n == 128) v = rbW[k];
      else if (n == 129) v = rgW[k];
      else if (n < 258)  v = wkW[(size_t)(n - 130) * 899 + k];
      else if (n == 258) v = wbW[k];
      else if (n == 259) v = wgW[k];
      else if (n < 388)  v = erW[(size_t)(n - 260) * 899 + k];
      else if (n < 516)  v = adW[(size_t)(n - 388) * 899 + k];
    }
    HWb[idx] = f2bf(v);
  }
  if (idx < 576) {
    int n = (int)idx; float bv = 0.f;
    if      (n < 128)  bv = rkb[n];
    else if (n == 128) bv = rbb[0];
    else if (n == 129) bv = rgb[0];
    else if (n < 258)  bv = wkb[n - 130];
    else if (n == 258) bv = wbb[0];
    else if (n == 259) bv = wgb[0];
    else if (n < 388)  bv = erb[n - 260];
    else if (n < 516)  bv = adb[n - 388];
    hbias[idx] = bv;
  }
}

__global__ void stage_pw(const float* pW, unsigned short* pWb)
{
  size_t idx = (size_t)blockIdx.x * blockDim.x + threadIdx.x;
  if (idx < (size_t)512 * 512) pWb[idx] = f2bf(pW[idx]);
}

__global__ void init_state(const float* mem_in, float* mem, float* rw,
                           float* h, float* c, float* rv)
{
  size_t idx = (size_t)blockIdx.x * blockDim.x + threadIdx.x;
  const size_t total = (size_t)16384 * 128;    // == 128*16384
  if (idx < total) { mem[idx] = mem_in[idx]; rw[idx] = 1.f / 16384.f; }
  if (idx < (size_t)128 * 1024) { h[idx] = 0.f; c[idx] = 0.f; }
  if (idx < (size_t)128 * 128)  rv[idx] = 0.f;
}

// ---------------------------------------------------------------------------
// Per-step kernels
// ---------------------------------------------------------------------------
__global__ void prep_input(const float* x, const float* rv, const float* h,
                           unsigned short* xrvh, int s)
{
  size_t idx = (size_t)blockIdx.x * blockDim.x + threadIdx.x;
  const size_t total = (size_t)128 * 1664;
  if (idx >= total) return;
  int b = (int)(idx / 1664), k = (int)(idx % 1664);
  float v;
  if      (k < 512)  v = x[((size_t)b * 32 + s) * 512 + k];
  else if (k < 640)  v = rv[(size_t)b * 128 + (k - 512)];
  else               v = h [(size_t)b * 1024 + (k - 640)];
  xrvh[idx] = f2bf(v);
}

__global__ void lstm_pointwise(const float* g, float* h, float* c,
                               unsigned short* hbf)
{
  size_t idx = (size_t)blockIdx.x * blockDim.x + threadIdx.x;
  if (idx >= (size_t)128 * 1024) return;
  int b = (int)(idx >> 10), j = (int)(idx & 1023);
  const float* gb = g + (size_t)b * 4096;
  float iv = gb[j], fv = gb[1024 + j], gv = gb[2048 + j], ov = gb[3072 + j];
  float cn = sigf(fv) * c[idx] + sigf(iv) * tanhf(gv);
  float hn = sigf(ov) * tanhf(cn);
  c[idx] = cn; h[idx] = hn;
  hbf[idx] = f2bf(hn);
}

__global__ void co_to_bf(const float* cof, unsigned short* cobf)
{
  size_t idx = (size_t)blockIdx.x * blockDim.x + threadIdx.x;
  if (idx >= (size_t)128 * 928) return;
  int b = (int)(idx / 928), k = (int)(idx % 928);
  cobf[idx] = f2bf(k < 899 ? cof[(size_t)b * 960 + k] : 0.f);
}

__global__ void head_pointwise(const float* hout, unsigned short* keybf,
                               float* rbeta, float* rgamma,
                               float* wbeta, float* wgamma,
                               float* er, float* ad)
{
  __shared__ float red[128];
  const int b = blockIdx.x, w = threadIdx.x;
  const float* hb = hout + (size_t)b * 576;
  float rk = hb[w], wk = hb[130 + w];
  red[w] = rk * rk; __syncthreads();
  for (int off = 64; off; off >>= 1) { if (w < off) red[w] += red[w + off]; __syncthreads(); }
  float rden = fmaxf(sqrtf(red[0]), 1e-12f);
  __syncthreads();
  red[w] = wk * wk; __syncthreads();
  for (int off = 64; off; off >>= 1) { if (w < off) red[w] += red[w + off]; __syncthreads(); }
  float wden = fmaxf(sqrtf(red[0]), 1e-12f);
  keybf[(size_t)b * 128 + w]         = f2bf(rk / rden);
  keybf[(size_t)(128 + b) * 128 + w] = f2bf(wk / wden);
  er[(size_t)b * 128 + w] = sigf(hb[260 + w]);
  ad[(size_t)b * 128 + w] = tanhf(hb[388 + w]);
  if (w == 0) {
    rbeta[b]  = softplusf(hb[128]);
    rgamma[b] = 1.f + softplusf(hb[129]);
    wbeta[b]  = softplusf(hb[258]);
    wgamma[b] = 1.f + softplusf(hb[259]);
  }
}

__global__ void mem_norm(const float* mem, unsigned short* memn)
{
  __shared__ float red[128];
  const int n = blockIdx.x, w = threadIdx.x;
  float v = mem[(size_t)n * 128 + w];
  red[w] = v * v; __syncthreads();
  for (int off = 64; off; off >>= 1) { if (w < off) red[w] += red[w + off]; __syncthreads(); }
  float den = fmaxf(sqrtf(red[0]), 1e-12f);
  memn[(size_t)n * 128 + w] = f2bf(v / den);
}

__global__ __launch_bounds__(256)
void softmax_interp(const float* simf,
                    const float* rbeta, const float* rgamma,
                    const float* wbeta, const float* wgamma,
                    float* rw, float* ww)
{
  __shared__ float red[256];
  const int row = blockIdx.x, t = threadIdx.x;
  const int b = row & 127;
  const bool isRead = row < 128;
  const float beta  = isRead ? rbeta[b]  : wbeta[b];
  const float gamma = isRead ? rgamma[b] : wgamma[b];
  const float* srow = simf + (size_t)row * 16384;

  float m = -3.4e38f;
  for (int n = t; n < 16384; n += 256) m = fmaxf(m, beta * srow[n]);
  red[t] = m; __syncthreads();
  for (int off = 128; off; off >>= 1) { if (t < off) red[t] = fmaxf(red[t], red[t + off]); __syncthreads(); }
  m = red[0]; __syncthreads();

  float s = 0.f;
  for (int n = t; n < 16384; n += 256) s += __expf(beta * srow[n] - m);
  red[t] = s; __syncthreads();
  for (int off = 128; off; off >>= 1) { if (t < off) red[t] += red[t + off]; __syncthreads(); }
  const float inv = 1.f / red[0];

  float* dst = isRead ? (rw + (size_t)b * 16384) : (ww + (size_t)b * 16384);
  const float uprev = 1.f / 16384.f;
  for (int n = t; n < 16384; n += 256) {
    float wv = __expf(beta * srow[n] - m) * inv;
    float prev = isRead ? dst[n] : uprev;
    dst[n] = gamma * wv + (1.f - gamma) * prev;
  }
}

__global__ void mem_update(const float* ww, const float* er, const float* ad,
                           float* mem)
{
  const int n = blockIdx.x, w = threadIdx.x;
  float ae = 0.f, aa = 0.f;
#pragma unroll 4
  for (int b = 0; b < 128; ++b) {
    float wv = ww[(size_t)b * 16384 + n];   // L2 resident
    ae = fmaf(wv, er[b * 128 + w], ae);
    aa = fmaf(wv, ad[b * 128 + w], aa);
  }
  size_t i = (size_t)n * 128 + w;
  mem[i] = mem[i] * (1.f - ae) + aa;
}

__global__ void read_vec(const float* rw, const float* mem, float* rv)
{
  const int b = blockIdx.x, w = threadIdx.x;
  const float* rwb = rw + (size_t)b * 16384;
  float acc = 0.f;
#pragma unroll 4
  for (int n = 0; n < 16384; ++n)
    acc = fmaf(rwb[n], mem[(size_t)n * 128 + w], acc);
  rv[(size_t)b * 128 + w] = acc;
}

// ---------------------------------------------------------------------------
// Launcher
// ---------------------------------------------------------------------------
extern "C" void kernel_launch(void* const* d_in, const int* in_sizes, int n_in,
                              void* d_out, int out_size, void* d_ws, size_t ws_size,
                              hipStream_t stream)
{
  (void)in_sizes; (void)n_in; (void)out_size; (void)ws_size;
  const float* x      = (const float*)d_in[0];
  const float* mem_in = (const float*)d_in[1];
  const float* Wih    = (const float*)d_in[2];
  const float* Whh    = (const float*)d_in[3];
  const float* bih    = (const float*)d_in[4];
  const float* bhh    = (const float*)d_in[5];
  const float* Wout   = (const float*)d_in[6];
  const float* bout   = (const float*)d_in[7];
  const float* pW     = (const float*)d_in[24];
  const float* pb     = (const float*)d_in[25];
  float* out = (float*)d_out;

  char* p = (char*)d_ws;
  auto alloc = [&](size_t bytes) -> void* {
    void* r = (void*)p;
    p += (bytes + 255) & ~(size_t)255;
    return r;
  };

  unsigned short* Wcat  = (unsigned short*)alloc((size_t)4096 * 1664 * 2);
  float*          biasg = (float*)alloc(4096 * 4);
  unsigned short* Woutb = (unsigned short*)alloc((size_t)960 * 1024 * 2);
  float*          biasc = (float*)alloc(960 * 4);
  unsigned short* HWb   = (unsigned short*)alloc((size_t)576 * 928 * 2);
  float*          biash = (float*)alloc(576 * 4);
  unsigned short* pWb   = (unsigned short*)alloc((size_t)512 * 512 * 2);
  float*          mem   = (float*)alloc((size_t)16384 * 128 * 4);
  unsigned short* memn  = (unsigned short*)alloc((size_t)16384 * 128 * 2);
  float*          g     = (float*)alloc((size_t)128 * 4096 * 4);
  float*          h     = (float*)alloc((size_t)128 * 1024 * 4);
  float*          c     = (float*)alloc((size_t)128 * 1024 * 4);
  unsigned short* hbf   = (unsigned short*)alloc((size_t)128 * 1024 * 2);
  float*          cof   = (float*)alloc((size_t)128 * 960 * 4);
  unsigned short* cobf  = (unsigned short*)alloc((size_t)128 * 928 * 2);
  float*          hout  = (float*)alloc((size_t)128 * 576 * 4);
  unsigned short* keybf = (unsigned short*)alloc((size_t)256 * 128 * 2);
  float*          simf  = (float*)alloc((size_t)256 * 16384 * 4);
  float*          rw    = (float*)alloc((size_t)128 * 16384 * 4);
  float*          ww    = (float*)alloc((size_t)128 * 16384 * 4);
  float*          rv    = (float*)alloc((size_t)128 * 128 * 4);
  float*          er    = (float*)alloc((size_t)128 * 128 * 4);
  float*          ad    = (float*)alloc((size_t)128 * 128 * 4);
  float*          rbeta = (float*)alloc(128 * 4);
  float*          rgamma= (float*)alloc(128 * 4);
  float*          wbeta = (float*)alloc(128 * 4);
  float*          wgamma= (float*)alloc(128 * 4);
  unsigned short* xrvh  = (unsigned short*)alloc((size_t)128 * 1664 * 2);

  auto blocks = [](size_t n) { return (unsigned)((n + 255) / 256); };

  // --- stage constant weights to bf16 (padded) ---
  stage_gates<<<blocks((size_t)4096 * 1664), 256, 0, stream>>>(Wih, Whh, bih, bhh, Wcat, biasg);
  stage_wout <<<blocks((size_t)960 * 1024), 256, 0, stream>>>(Wout, bout, Woutb, biasc);
  stage_heads<<<blocks((size_t)576 * 928), 256, 0, stream>>>(
      (const float*)d_in[8],  (const float*)d_in[9],
      (const float*)d_in[10], (const float*)d_in[11],
      (const float*)d_in[12], (const float*)d_in[13],
      (const float*)d_in[14], (const float*)d_in[15],
      (const float*)d_in[16], (const float*)d_in[17],
      (const float*)d_in[18], (const float*)d_in[19],
      (const float*)d_in[20], (const float*)d_in[21],
      (const float*)d_in[22], (const float*)d_in[23],
      HWb, biash);
  stage_pw<<<blocks((size_t)512 * 512), 256, 0, stream>>>(pW, pWb);
  init_state<<<blocks((size_t)16384 * 128), 256, 0, stream>>>(mem_in, mem, rw, h, c, rv);

  // --- sequential scan over S=32 timesteps ---
  for (int s = 0; s < 32; ++s) {
    prep_input<<<blocks((size_t)128 * 1664), 256, 0, stream>>>(x, rv, h, xrvh, s);

    // g = [x|rv|h] @ [Wih|Whh]^T + (bih+bhh) : M=128 N=4096 K=1664
    gemm_bf16_nt<<<dim3(64, 1), 256, 0, stream>>>(xrvh, Wcat, biasg, g,
                                                  128, 4096, 1664, 1664, 1664, 4096);
    lstm_pointwise<<<blocks((size_t)128 * 1024), 256, 0, stream>>>(g, h, c, hbf);

    // co = h @ Wout^T + bout : M=128 N=960(pad 899) K=1024
    gemm_bf16_nt<<<dim3(15, 1), 256, 0, stream>>>(hbf, Woutb, biasc, cof,
                                                  128, 960, 1024, 1024, 1024, 960);
    co_to_bf<<<blocks((size_t)128 * 928), 256, 0, stream>>>(cof, cobf);

    // all head projections fused : M=128 N=576(pad 516) K=928(pad 899)
    gemm_bf16_nt<<<dim3(9, 1), 256, 0, stream>>>(cobf, HWb, biash, hout,
                                                 128, 576, 928, 928, 928, 576);
    head_pointwise<<<128, 128, 0, stream>>>(hout, keybf, rbeta, rgamma,
                                            wbeta, wgamma, er, ad);

    // cosine similarity (read+write keys stacked): M=256 N=16384 K=128
    mem_norm<<<16384, 128, 0, stream>>>(mem, memn);
    gemm_bf16_nt<<<dim3(256, 2), 256, 0, stream>>>(keybf, memn, nullptr, simf,
                                                   256, 16384, 128, 128, 128, 16384);
    softmax_interp<<<256, 256, 0, stream>>>(simf, rbeta, rgamma, wbeta, wgamma, rw, ww);

    // memory write (erase/add summed over batch), then read vector
    mem_update<<<16384, 128, 0, stream>>>(ww, er, ad, mem);
    read_vec<<<128, 128, 0, stream>>>(rw, mem, rv);

    // out[:, s, :] = co[:, :512] @ pW^T + pb : M=128 N=512 K=512 (lda=928 strided)
    gemm_bf16_nt<<<dim3(8, 1), 256, 0, stream>>>(cobf, pWb, pb, out + (size_t)s * 512,
                                                 128, 512, 512, 928, 512, 32 * 512);
  }
}
